// ExtractModel_80376017977802
// MI455X (gfx1250) — matrix operand order repaired and checked by reading, not compiled, not run on hardware
//
#include <hip/hip_runtime.h>
#include <hip/hip_bf16.h>
#include <math.h>

// ---------------- problem constants (match reference) ----------------
constexpr int   cV     = 1000;
constexpr int   cL     = 10;
constexpr int   cLOST  = 2000;
constexpr int   cKNOWN = 400;
constexpr int   cDIM   = 256;
constexpr int   cB     = 8;
constexpr int   cS     = 256;
constexpr int   cKW    = 3;
constexpr float cCW    = 0.2f;
constexpr float cIDC   = 3.5f;
constexpr float cNEG   = -1000000000.0f;

constexpr int   cLOSTP = 2048;   // lost_unit_emb padded rows (zeros beyond 2000)
constexpr int   cKPAD  = 416;    // aligner K padded 400 -> 416 (multiple of 32)

typedef __attribute__((ext_vector_type(16))) _Float16 v16h;
typedef __attribute__((ext_vector_type(8)))  float    v8f;

// ---------------- Tensor Data Mover (gfx1250) support ----------------
#if defined(__gfx1250__) && __has_builtin(__builtin_amdgcn_tensor_load_to_lds)
#define HAVE_TDM 1
#if __has_include(<hip/amd_detail/amd_gfx1250_TDM.h>)
#define TDM_ARITY6 1       // therock headers -> 6-arg builtin
#else
#define TDM_ARITY6 0       // ROCm 7.2 -> 5-arg builtin
#endif
#else
#define HAVE_TDM 0
#endif

#if HAVE_TDM
typedef unsigned int u32x4 __attribute__((ext_vector_type(4)));
typedef int          i32x4 __attribute__((ext_vector_type(4)));
typedef int          i32x8 __attribute__((ext_vector_type(8)));

// 2-D f32 tile load, global -> LDS, via TDM descriptor (D# groups 0/1).
// tile_c = contiguous elements per row, tile_r = rows, stride = row pitch (elems).
// tensor dims are set equal to the tile so no OOB clipping is involved:
// every call site guarantees the tile is fully in-bounds.
__device__ __forceinline__ void tdm_load_2d(void* lds_dst, const void* gsrc,
                                            unsigned tile_c, unsigned tile_r,
                                            unsigned stride_elems) {
  const unsigned lds_byte = (unsigned)(uintptr_t)lds_dst;        // LDS offset
  const unsigned long long ga = (unsigned long long)(uintptr_t)gsrc;
  u32x4 g0;
  g0[0] = 1u;                                           // count=1, user mode
  g0[1] = lds_byte;                                     // lds_addr (bytes)
  g0[2] = (unsigned)ga;                                 // global_addr[31:0]
  g0[3] = (unsigned)((ga >> 32) & 0x01FFFFFFu) | (2u << 30);  // [56:32] | type=2
  const unsigned long long st0 = stride_elems;          // 48-bit stride (elems)
  i32x8 g1;
  g1[0] = (int)(2u << 16);                              // data_size = 4 bytes
  g1[1] = (int)((tile_c & 0xFFFFu) << 16);              // tensor_dim0[15:0]
  g1[2] = (int)((tile_c >> 16) | ((tile_r & 0xFFFFu) << 16)); // td0 hi | td1 lo
  g1[3] = (int)((tile_r >> 16) | (tile_c << 16));       // td1 hi | tile_dim0
  g1[4] = (int)(tile_r & 0xFFFFu);                      // tile_dim1 (tile_dim2=0)
  g1[5] = (int)(unsigned)st0;                           // dim0 stride [31:0]
  g1[6] = (int)(unsigned)((st0 >> 32) & 0xFFFFu);       // dim0 stride [47:32]
  g1[7] = 0;
  const i32x4 z4 = {0, 0, 0, 0};
#if TDM_ARITY6
  const i32x8 z8 = {0, 0, 0, 0, 0, 0, 0, 0};
  __builtin_amdgcn_tensor_load_to_lds(g0, g1, z4, z4, z8, 0);
#else
  __builtin_amdgcn_tensor_load_to_lds(g0, g1, z4, z4, 0);
#endif
}
#endif  // HAVE_TDM

// ---------------------------------------------------------------------
// LDS-staged WMMA GEMM: C(M,N) = A(M,K) * B(K,N) [+ bias(N)].
// TRANSB: B stored row-major as (N,K) -> computes A * B^T.
// Block: 256 threads = 8 waves. Block tile: 16(M) x 128(N); wave tile 16x16.
// Per 32-wide K chunk, wave 0 DMA's the A/B panels into LDS with the TDM,
// waits on TENSORcnt, barrier; all waves then feed v_wmma_f32_16x16x32_f16
// from LDS. Requires: M%16==0, K%32==0 (call sites pad), and for the
// ragged-N GEMMs the B buffer is row-padded so panels stay in-bounds.
// ---------------------------------------------------------------------
template <bool TRANSB>
__global__ void gemm_tdm_wmma(const float* __restrict__ A,
                              const float* __restrict__ B,
                              float* __restrict__ C,
                              const float* __restrict__ bias,
                              int M, int N, int K) {
  constexpr int BN = 128;
  __shared__ float Alds[16 * 32];          // [m][k], stride 32
  __shared__ float Blds[32 * BN];          // TRANSB: [n][k] stride 32; else [k][n] stride BN
  const int wave = threadIdx.x >> 5;
  const int lane = threadIdx.x & 31;
  const int NT    = N >> 4;
  const int nblkN = (NT + 7) >> 3;
  const int bm  = blockIdx.x / nblkN;      // M tile (16 rows)
  const int bn0 = (blockIdx.x % nblkN) * 8;
  const int tn  = bn0 + wave;              // this wave's N tile
  const int half = lane >> 4;
  const int mr   = lane & 15;

  v8f acc = {};
  const int nchunks = K >> 5;
  for (int kc = 0; kc < nchunks; ++kc) {
    const int k0 = kc << 5;
    __syncthreads();                       // previous chunk fully consumed
#if HAVE_TDM
    if (wave == 0) {
      tdm_load_2d(Alds, A + (size_t)bm * 16 * K + k0, 32u, 16u, (unsigned)K);
      if (TRANSB) {   // B (N,K): rows n = bn0*16 .. +128, cols k0..k0+31
        tdm_load_2d(Blds, B + (size_t)(bn0 * 16) * K + k0, 32u, (unsigned)BN,
                    (unsigned)K);
      } else {        // B (K,N): rows k0..k0+31, cols bn0*16 .. +128
        tdm_load_2d(Blds, B + (size_t)k0 * N + bn0 * 16, (unsigned)BN, 32u,
                    (unsigned)N);
      }
      __builtin_amdgcn_s_wait_tensorcnt(0);
    }
#else
    for (int i = threadIdx.x; i < 16 * 32; i += blockDim.x)
      Alds[i] = A[(size_t)(bm * 16 + (i >> 5)) * K + k0 + (i & 31)];
    if (TRANSB) {
      for (int i = threadIdx.x; i < 32 * BN; i += blockDim.x)
        Blds[i] = B[(size_t)(bn0 * 16 + (i >> 5)) * K + k0 + (i & 31)];
    } else {
      for (int i = threadIdx.x; i < 32 * BN; i += blockDim.x)
        Blds[i] = B[(size_t)(k0 + i / BN) * N + bn0 * 16 + (i % BN)];
    }
#endif
    __syncthreads();

    // A fragment (16x32 f16, this kernel's contiguous k order): lane half h -> K = 16h+{0..15}
    v16h a, b;
    const float* ar = &Alds[mr * 32 + 16 * half];
#pragma unroll
    for (int e = 0; e < 8; ++e) a[e]     = (_Float16)ar[e];
#pragma unroll
    for (int e = 0; e < 8; ++e) a[8 + e] = (_Float16)ar[8 + e];
    // B fragment (32x16 f16, same contiguous k order as A): lane half h -> K = 16h+{0..15}, col = lane&15
    if (TRANSB) {
      const float* br = &Blds[(wave * 16 + mr) * 32 + 16 * half];
#pragma unroll
      for (int e = 0; e < 16; ++e) b[e] = (_Float16)br[e];
    } else {
      const float* br = &Blds[(16 * half) * BN + wave * 16 + mr];
#pragma unroll
      for (int e = 0; e < 16; ++e) b[e] = (_Float16)br[e * BN];
    }
    acc = __builtin_amdgcn_wmma_f32_16x16x32_f16(
        false, a, false, b, (short)0, acc, false, false);
  }

  if (tn < NT) {                           // ragged-N guard (after all barriers/wmma)
    const int   ccol = tn * 16 + mr;
    const float bv   = bias ? bias[ccol] : 0.0f;
#pragma unroll
    for (int r = 0; r < 8; ++r)
      C[(size_t)(bm * 16 + r + 8 * half) * N + ccol] = acc[r] + bv;
  }
}

// Zero-padded 2-D copy: out(Mout,Kout) <- in(Min,Kin)
__global__ void pad2d_kernel(const float* __restrict__ in, float* __restrict__ out,
                             int Min, int Kin, int Mout, int Kout) {
  const int idx = blockIdx.x * blockDim.x + threadIdx.x;
  if (idx >= Mout * Kout) return;
  const int r = idx / Kout, c = idx % Kout;
  out[idx] = (r < Min && c < Kin) ? in[(size_t)r * Kin + c] : 0.0f;
}

__global__ void zero_kernel(float* __restrict__ p, int n) {
  const int i = blockIdx.x * blockDim.x + threadIdx.x;
  if (i < n) p[i] = 0.0f;
}

// Pack conv_w (DIM_out, DIM_in, KW) -> W2 (KW*DIM_in, DIM_out)
__global__ void prep_w2_kernel(const float* __restrict__ conv_w,
                               float* __restrict__ W2) {
  const int idx = blockIdx.x * blockDim.x + threadIdx.x;
  if (idx >= cKW * cDIM * cDIM) return;
  const int d = idx & (cDIM - 1);
  const int c = idx >> 8;          // c = k*DIM + i
  const int k = c >> 8;
  const int i = c & (cDIM - 1);
  W2[idx] = conv_w[((size_t)d * cDIM + i) * cKW + k];
}

// im2col: X[(v*L+l), k*DIM+i] = masked kv_emb[v, l+k-1, i] (SAME pad, length mask)
__global__ void im2col_kernel(const int* __restrict__ vocab_ids,
                              const int* __restrict__ vocab_length,
                              const float* __restrict__ known_emb,
                              float* __restrict__ X) {
  const int idx = blockIdx.x * blockDim.x + threadIdx.x;
  if (idx >= cV * cL * cKW * cDIM) return;
  const int i    = idx & (cDIM - 1);
  const int rest = idx >> 8;
  const int kk   = rest % cKW;
  const int row  = rest / cKW;     // v*L + l
  const int v    = row / cL;
  const int l    = row % cL;
  const int lsrc = l + kk - 1;
  float val = 0.0f;
  if (lsrc >= 0 && lsrc < cL && lsrc < vocab_length[v]) {
    val = known_emb[(size_t)vocab_ids[v * cL + lsrc] * cDIM + i];
  }
  X[idx] = val;
}

// In-place row log-softmax: one block per row
__global__ void row_lsm_kernel(float* __restrict__ data, int ncols) {
  float* p = data + (size_t)blockIdx.x * ncols;
  __shared__ float red[256];
  const int tid = threadIdx.x;
  float m = -INFINITY;
  for (int c = tid; c < ncols; c += blockDim.x) m = fmaxf(m, p[c]);
  red[tid] = m; __syncthreads();
  for (int st = blockDim.x >> 1; st; st >>= 1) {
    if (tid < st) red[tid] = fmaxf(red[tid], red[tid + st]);
    __syncthreads();
  }
  m = red[0]; __syncthreads();
  float sum = 0.0f;
  for (int c = tid; c < ncols; c += blockDim.x) sum += __expf(p[c] - m);
  red[tid] = sum; __syncthreads();
  for (int st = blockDim.x >> 1; st; st >>= 1) {
    if (tid < st) red[tid] += red[tid + st];
    __syncthreads();
  }
  const float lse = m + __logf(red[0]);
  for (int c = tid; c < ncols; c += blockDim.x) p[c] -= lse;
}

// Fused: sub = cw*log_softmax(scores) + (1-cw)*ulp[vocab_ids[row]]  (in place)
__global__ void sub_fuse_kernel(float* __restrict__ scores,
                                const float* __restrict__ ulp,
                                const int* __restrict__ vocab_ids,
                                int ncols) {
  const int rid = blockIdx.x;                // rid = v*L + l
  float* p = scores + (size_t)rid * ncols;
  const float* g = ulp + (size_t)vocab_ids[rid] * ncols;
  __shared__ float red[256];
  const int tid = threadIdx.x;
  float m = -INFINITY;
  for (int c = tid; c < ncols; c += blockDim.x) m = fmaxf(m, p[c]);
  red[tid] = m; __syncthreads();
  for (int st = blockDim.x >> 1; st; st >>= 1) {
    if (tid < st) red[tid] = fmaxf(red[tid], red[tid + st]);
    __syncthreads();
  }
  m = red[0]; __syncthreads();
  float sum = 0.0f;
  for (int c = tid; c < ncols; c += blockDim.x) sum += __expf(p[c] - m);
  red[tid] = sum; __syncthreads();
  for (int st = blockDim.x >> 1; st; st >>= 1) {
    if (tid < st) red[tid] += red[tid + st];
    __syncthreads();
  }
  const float lse = m + __logf(red[0]);
  for (int c = tid; c < ncols; c += blockDim.x)
    p[c] = cCW * (p[c] - lse) + (1.0f - cCW) * g[c];
}

// span_lp[k-1,b,s] = LSE over {v: len_v==k} of sum_{j<k} sub[v,j,tok(b,s+j)] - logV
// One wave per (b,s); lanes stripe over v; 8 waves/block share one b row.
__global__ void span_kernel(const float* __restrict__ sub,
                            const int* __restrict__ text_ids,
                            const int* __restrict__ vocab_length,
                            float* __restrict__ span) {
  __shared__ int sh_len[cV];
  __shared__ int sh_tok[cS];
  const int wavesPerBlock = blockDim.x >> 5;      // 8
  const int p0 = blockIdx.x * wavesPerBlock;
  const int b  = p0 / cS;                         // uniform across block
  for (int i = threadIdx.x; i < cV; i += blockDim.x) sh_len[i] = vocab_length[i];
  for (int i = threadIdx.x; i < cS; i += blockDim.x) sh_tok[i] = text_ids[b * cS + i];
  __syncthreads();

  const int wave = threadIdx.x >> 5;
  const int lane = threadIdx.x & 31;
  const int s    = (p0 + wave) % cS;
  const float logV = __logf((float)cV);

  for (int k = 1; k <= cL; ++k) {
    float m = cNEG, ssum = 0.0f;
    for (int v = lane; v < cV; v += 32) {
      if (sh_len[v] != k) continue;
      float c = 0.0f;
      for (int j = 0; j < k; ++j) {
        const int sp = s + j;
        c += (sp < cS) ? sub[(size_t)(v * cL + j) * cLOST + sh_tok[sp]] : cNEG;
      }
      const float mn = fmaxf(m, c);
      ssum = ssum * __expf(m - mn) + __expf(c - mn);
      m = mn;
    }
    for (int off = 16; off; off >>= 1) {        // wave32 LSE reduction
      const float mo = __shfl_xor(m, off, 32);
      const float so = __shfl_xor(ssum, off, 32);
      const float mn = fmaxf(m, mo);
      ssum = ssum * __expf(m - mn) + so * __expf(mo - mn);
      m = mn;
    }
    if (lane == 0) {
      const float val = (ssum > 0.0f) ? (m + __logf(ssum) - logV) : cNEG;
      span[((size_t)(k - 1) * cB + b) * cS + s] = val;
    }
  }
}

// Sequential DP over S; lane b (< B) owns one batch row.
__global__ void dp_kernel(const float* __restrict__ span,
                          float* __restrict__ out) {
  const int b = threadIdx.x;
  if (b >= cB) return;
  float w[cL];
#pragma unroll
  for (int i = 0; i < cL; ++i) w[i] = cNEG;
  w[cL - 1] = 0.0f;
  for (int t = 0; t < cS; ++t) {
    float vals[cL + 1];
    vals[0] = w[cL - 1] - cIDC;       // skip
#pragma unroll
    for (int k = 1; k <= cL; ++k) {
      const int ss = t - k + 1;
      vals[k] = (ss >= 0)
                  ? (w[cL - k] + span[((size_t)(k - 1) * cB + b) * cS + ss])
                  : cNEG;
    }
    float m = vals[0];
#pragma unroll
    for (int k = 1; k <= cL; ++k) m = fmaxf(m, vals[k]);
    float ssum = 0.0f;
#pragma unroll
    for (int k = 0; k <= cL; ++k) ssum += __expf(vals[k] - m);
    const float node = m + __logf(ssum);
#pragma unroll
    for (int i = 0; i < cL - 1; ++i) w[i] = w[i + 1];
    w[cL - 1] = node;
    out[(size_t)b * cS + t] = node;
  }
}

// ---------------------------------------------------------------------
extern "C" void kernel_launch(void* const* d_in, const int* in_sizes, int n_in,
                              void* d_out, int out_size, void* d_ws, size_t ws_size,
                              hipStream_t stream) {
  (void)in_sizes; (void)n_in; (void)out_size; (void)ws_size;
  const int*   vocab_ids    = (const int*)d_in[0];
  const int*   vocab_length = (const int*)d_in[1];
  const int*   text_ids     = (const int*)d_in[2];
  // d_in[3] text_lengths: unused by reference
  const float* aligner      = (const float*)d_in[4];
  const float* known        = (const float*)d_in[5];
  const float* conv_w       = (const float*)d_in[6];
  const float* conv_b       = (const float*)d_in[7];
  // d_in[8..9] ins_conv_w/b: dead in reference (ins/alignment unused)
  float* out = (float*)d_out;

  // workspace layout (floats); im2col X overlays the scores/sub region
  float* ws    = (float*)d_ws;
  float* W2    = ws;                                     // 196608
  float* lostP = W2    + (size_t)cKW * cDIM * cDIM;      // 2048*256 = 524288
  float* ulp   = lostP + (size_t)cLOSTP * cDIM;          // 400*2000 = 800000
  float* ctx   = ulp   + (size_t)cKNOWN * cLOST;         // 10000*256 = 2560000
  float* spanb = ctx   + (size_t)cV * cL * cDIM;         // 20480
  float* Apad  = spanb + (size_t)cL * cB * cS;           // 2000*416 = 832000
  float* Bpad  = Apad  + (size_t)cLOST * cKPAD;          // 416*256 = 106496
  float* subX  = Bpad  + (size_t)cKPAD * cDIM;           // 20000000
  // total ~25.04M floats (~100 MB)

  // 0) prep: weight pack, im2col (X in subX region), K/row padding, lost tail zero
  prep_w2_kernel<<<(cKW * cDIM * cDIM + 255) / 256, 256, 0, stream>>>(conv_w, W2);
  im2col_kernel<<<(cV * cL * cKW * cDIM + 255) / 256, 256, 0, stream>>>(
      vocab_ids, vocab_length, known, subX);
  pad2d_kernel<<<(cLOST * cKPAD + 255) / 256, 256, 0, stream>>>(
      aligner, Apad, cLOST, cKNOWN, cLOST, cKPAD);            // (2000,400)->(2000,416)
  pad2d_kernel<<<(cKPAD * cDIM + 255) / 256, 256, 0, stream>>>(
      known, Bpad, cKNOWN, cDIM, cKPAD, cDIM);                // (400,256)->(416,256)
  zero_kernel<<<((cLOSTP - cLOST) * cDIM + 255) / 256, 256, 0, stream>>>(
      lostP + (size_t)cLOST * cDIM, (cLOSTP - cLOST) * cDIM); // pad rows of lost

  // 1) lost = Apad(2000x416) @ Bpad(416x256)
  {
    const int grid = (cLOST / 16) * (((cDIM / 16) + 7) / 8);       // 125*2
    gemm_tdm_wmma<false><<<grid, 256, 0, stream>>>(
        Apad, Bpad, lostP, nullptr, cLOST, cDIM, cKPAD);
  }
  // 2) logits = known(400x256) @ lost^T -> ulp; then row log-softmax
  {
    const int grid = (cKNOWN / 16) * (((cLOST / 16) + 7) / 8);     // 25*16
    gemm_tdm_wmma<true><<<grid, 256, 0, stream>>>(
        known, lostP, ulp, nullptr, cKNOWN, cLOST, cDIM);
  }
  row_lsm_kernel<<<cKNOWN, 256, 0, stream>>>(ulp, cLOST);

  // 3) ctx = X(10000x768) @ W2(768x256) + conv_b   (the conv1d)
  {
    const int grid = (cV * cL / 16) * (((cDIM / 16) + 7) / 8);     // 625*2
    gemm_tdm_wmma<false><<<grid, 256, 0, stream>>>(
        subX, W2, ctx, conv_b, cV * cL, cDIM, cKW * cDIM);
  }
  // 4) scores = ctx(10000x256) @ lost^T -> subX (overwrites consumed X)
  {
    const int grid = (cV * cL / 16) * (((cLOST / 16) + 7) / 8);    // 625*16
    gemm_tdm_wmma<true><<<grid, 256, 0, stream>>>(
        ctx, lostP, subX, nullptr, cV * cL, cLOST, cDIM);
  }
  // 5) sub = 0.2*log_softmax(scores) + 0.8*ulp[vocab_ids]  (in place)
  sub_fuse_kernel<<<cV * cL, 256, 0, stream>>>(subX, ulp, vocab_ids, cLOST);

  // 6) span_lp over (length bucket, b, s)
  span_kernel<<<(cB * cS) / 8, 256, 0, stream>>>(subX, text_ids, vocab_length, spanb);

  // 7) sequential DP -> final_nodes (B,S)
  dp_kernel<<<1, 32, 0, stream>>>(spanb, out);
}
